// HyenaFilter_67534065762947
// MI455X (gfx1250) — matrix operand order repaired and checked by reading, not compiled, not run on hardware
//
// Hyena filter + FFT long convolution for MI455X (gfx1250, wave32, WMMA).
//
//   Kernel 1 (hyena_filter_kernel): MLP filter generation; the dominant
//     (L x 64)@(64 x 1024) projection uses v_wmma_f32_16x16x32_f16 with Wout
//     staged transposed in LDS. Decay fused into epilogue; k stored
//     channel-major f16 in d_ws.
//   Kernel 2 (hyena_fftconv_kernel): one workgroup per channel. Four-step
//     matmul FFT (16384 = 128 x 128); each 128-pt DFT stage is a complex
//     GEMM of WMMA f16 tiles with f32 accumulation. All operand fetches are
//     ds_load_b128; B-fragments are hoisted per wave (fixed N-column), and
//     the inter-stage twiddle rotation is fused into the stage-1 store
//     epilogue. DFT matrix, filter spectrum, working set: 272 KB LDS.

#include <hip/hip_runtime.h>

typedef __attribute__((ext_vector_type(16))) _Float16 v16h;
typedef __attribute__((ext_vector_type(8)))  _Float16 v8h;
typedef __attribute__((ext_vector_type(8)))  float    v8f;

#define PI_F 3.14159265358979323846f
#define TW_STEP (2.0f * PI_F / 16384.0f)

static __device__ __forceinline__ v8f wmma_f16(v16h a, v16h b, v8f c) {
  return __builtin_amdgcn_wmma_f32_16x16x32_f16(false, a, false, b, (short)0, c,
                                                false, false);
}

// A-operand tile (16x32 f16) from row-major LDS: two ds_load_b128 per lane.
static __device__ __forceinline__ v16h load_a_frag(const _Float16* p, int ld,
                                                   int m0, int k0, int lane) {
  const _Float16* base =
      p + (m0 + (lane & 15)) * ld + k0 + ((lane >> 4) & 1) * 8;
  const v8h lo = *(const v8h*)base;
  const v8h hi = *(const v8h*)(base + 16);
  return __builtin_shufflevector(lo, hi, 0, 1, 2, 3, 4, 5, 6, 7, 8, 9, 10, 11,
                                 12, 13, 14, 15);
}

// B-operand tile (32x16 f16) from TRANSPOSED ([n][k]) LDS storage: two
// ds_load_b128 per lane.
static __device__ __forceinline__ v16h load_bt_frag(const _Float16* p, int ld,
                                                    int k0, int n0, int lane) {
  const _Float16* base =
      p + (n0 + (lane & 15)) * ld + k0 + ((lane >> 4) & 1) * 8;
  const v8h lo = *(const v8h*)base;
  const v8h hi = *(const v8h*)(base + 16);
  return __builtin_shufflevector(lo, hi, 0, 1, 2, 3, 4, 5, 6, 7, 8, 9, 10, 11,
                                 12, 13, 14, 15);
}

// Complex tiled GEMM across 8 waves; Ntiles fixed at 8 so each wave owns one
// N-column -> B fragments are hoisted into registers once per call.
//   C = (Pre + i*sp*Pim)*(Qre + i*sq*Qim); P row-major, Q transposed storage.
//   pim/qim/cim may be null. transC: store C as [col][row] (v8h stores).
//   twSign != 0: multiply C elementwise by exp(i*twSign*2pi*row*col/16384)
//   (fused inter-stage twiddle; index product is transpose-invariant).
template <int KT>
static __device__ __forceinline__ void wave_cgemm(
    const _Float16* pre, const _Float16* pim, int ldp,
    const _Float16* qre, const _Float16* qim, int ldq,
    _Float16* cre, _Float16* cim, int ldc, bool transC,
    int Mtiles, float sp, float sq, float scale, float twSign) {
  const int wave = (int)(threadIdx.x >> 5);
  const int lane = (int)(threadIdx.x & 31);
  const int nt   = wave;  // fixed column block per wave
  v16h bre[KT], bim[KT];
#pragma unroll
  for (int kt = 0; kt < KT; ++kt) {
    bre[kt] = load_bt_frag(qre, ldq, kt * 32, nt * 16, lane);
    if (qim) bim[kt] = load_bt_frag(qim, ldq, kt * 32, nt * 16, lane);
  }
  const int col = nt * 16 + (lane & 15);
  const int rbo = ((lane >> 4) & 1) << 3;
  const float ss = sp * sq;
  for (int mt = 0; mt < Mtiles; ++mt) {
    v8f arr = {}, aii = {}, ari = {}, air = {};
#pragma unroll
    for (int kt = 0; kt < KT; ++kt) {
      const v16h aRe = load_a_frag(pre, ldp, mt * 16, kt * 32, lane);
      arr = wmma_f16(aRe, bre[kt], arr);
      if (pim) {
        const v16h aIm = load_a_frag(pim, ldp, mt * 16, kt * 32, lane);
        if (cim) air = wmma_f16(aIm, bre[kt], air);
        if (qim) {
          aii = wmma_f16(aIm, bim[kt], aii);
          if (cim) ari = wmma_f16(aRe, bim[kt], ari);
        }
      } else if (qim && cim) {
        ari = wmma_f16(aRe, bim[kt], ari);
      }
    }
    const int rb = mt * 16 + rbo;
    v8h vre, vim;
#pragma unroll
    for (int v = 0; v < 8; ++v) {
      float re = (arr[v] - ss * aii[v]) * scale;
      float im = cim ? (sq * ari[v] + sp * air[v]) * scale : 0.f;
      if (twSign != 0.f) {
        float s, c;
        __sincosf(twSign * TW_STEP * (float)((rb + v) * col), &s, &c);
        const float r2 = re * c - im * s;
        im = re * s + im * c;
        re = r2;
      }
      vre[v] = (_Float16)re;
      vim[v] = (_Float16)im;
    }
    if (transC) {
      *(v8h*)(cre + col * ldc + rb) = vre;
      if (cim) *(v8h*)(cim + col * ldc + rb) = vim;
    } else {
#pragma unroll
      for (int v = 0; v < 8; ++v) {
        cre[(rb + v) * ldc + col] = vre[v];
        if (cim) cim[(rb + v) * ldc + col] = vim[v];
      }
    }
  }
}

// ---------------------------------------------------------------------------
// Kernel 1: filter generation. 64 blocks x 256 threads; block = 128 L-rows.
// ---------------------------------------------------------------------------
__global__ __launch_bounds__(256) void hyena_filter_kernel(
    const float* __restrict__ z, const float* __restrict__ tpos,
    const float* __restrict__ deltas, const float* __restrict__ freq,
    const float* __restrict__ W0, const float* __restrict__ b0,
    const float* __restrict__ W1, const float* __restrict__ b1,
    const float* __restrict__ W2, const float* __restrict__ b2,
    const float* __restrict__ Wout, _Float16* __restrict__ kf) {
  __shared__ float    sZ[128 * 33];     // 16.5 KB (z chunk, f32)
  __shared__ _Float16 sHA[128 * 64];    // 16 KB
  __shared__ _Float16 sHB[128 * 64];    // 16 KB
  __shared__ _Float16 sWt[1024 * 64];   // 128 KB (Wout transposed, f16)
  const int tid  = (int)threadIdx.x;
  const int row0 = (int)blockIdx.x * 128;

  for (int i = tid; i < 64 * 1024; i += 256) {
    const int k = i >> 10, n = i & 1023;
    sWt[n * 64 + k] = (_Float16)Wout[i];
  }
  for (int i = tid; i < 128 * 33; i += 256) sZ[i] = z[(size_t)row0 * 33 + i];
  __syncthreads();

  // Two threads per row; 32 units each; row inputs cached in registers.
  const int r  = tid >> 1;
  const int u0 = (tid & 1) * 32;
  {
    float zreg[33];
#pragma unroll
    for (int e = 0; e < 33; ++e) zreg[e] = sZ[r * 33 + e];
    for (int uu = 0; uu < 32; ++uu) {
      const int u = u0 + uu;
      float acc = b0[u];
#pragma unroll
      for (int e = 0; e < 33; ++e) acc += zreg[e] * W0[e * 64 + u];
      sHA[r * 64 + u] = (_Float16)__sinf(freq[u] * acc);
    }
  }
  __syncthreads();
  {
    v8h hv[8];
    const v8h* hrow = (const v8h*)(sHA + r * 64);
#pragma unroll
    for (int j = 0; j < 8; ++j) hv[j] = hrow[j];
    for (int uu = 0; uu < 32; ++uu) {
      const int u = u0 + uu;
      float acc = b1[u];
#pragma unroll
      for (int e = 0; e < 64; ++e)
        acc += (float)hv[e >> 3][e & 7] * W1[e * 64 + u];
      sHB[r * 64 + u] = (_Float16)__sinf(freq[u] * acc);
    }
  }
  __syncthreads();
  {
    v8h hv[8];
    const v8h* hrow = (const v8h*)(sHB + r * 64);
#pragma unroll
    for (int j = 0; j < 8; ++j) hv[j] = hrow[j];
    for (int uu = 0; uu < 32; ++uu) {
      const int u = u0 + uu;
      float acc = b2[u];
#pragma unroll
      for (int e = 0; e < 64; ++e)
        acc += (float)hv[e >> 3][e & 7] * W2[e * 64 + u];
      sHA[r * 64 + u] = (_Float16)__sinf(freq[u] * acc);
    }
  }
  __syncthreads();

  // (128x64)@(64x1024) WMMA. Each wave owns one M-tile -> A frags + tpos
  // hoisted out of the 64-column loop. Decay fused; store k channel-major.
  const int wave = tid >> 5, lane = tid & 31;
  const int mt = wave;
  const v16h a0 = load_a_frag(sHA, 64, mt * 16, 0, lane);
  const v16h a1 = load_a_frag(sHA, 64, mt * 16, 32, lane);
  const int rb = mt * 16 + (((lane >> 4) & 1) << 3);
  float tv[8];
#pragma unroll
  for (int v = 0; v < 8; ++v) tv[v] = tpos[row0 + rb + v];
  for (int nt = 0; nt < 64; ++nt) {
    v8f acc = {};
    acc = wmma_f16(a0, load_bt_frag(sWt, 64, 0, nt * 16, lane), acc);
    acc = wmma_f16(a1, load_bt_frag(sWt, 64, 32, nt * 16, lane), acc);
    const int col = nt * 16 + (lane & 15);  // channel d
    const float ad = fabsf(deltas[col]);
#pragma unroll
    for (int v = 0; v < 8; ++v) {
      const float scl = __expf(-tv[v] * ad) + 0.05f;
      kf[(size_t)col * 8192 + (row0 + rb + v)] = (_Float16)(acc[v] * scl);
    }
  }
}

// ---------------------------------------------------------------------------
// Kernel 2: FFT convolution. 1024 blocks (one per channel) x 256 threads.
// ---------------------------------------------------------------------------
__global__ __launch_bounds__(256) void hyena_fftconv_kernel(
    const float* __restrict__ x, const _Float16* __restrict__ kf,
    const float* __restrict__ Dvec, float* __restrict__ out) {
  __shared__ _Float16 sDre[128 * 128], sDim[128 * 128];  // DFT-128 (symmetric)
  __shared__ _Float16 sAt[128 * 64];                     // input, transposed
  __shared__ _Float16 sXre[128 * 128], sXim[128 * 128];  // spectrum / output
  __shared__ _Float16 sGre[128 * 128], sGim[128 * 128];  // stage buffer
  __shared__ _Float16 sKre[128 * 128], sKim[128 * 128];  // filter spectrum
  const int tid = (int)threadIdx.x;
  const int d   = (int)blockIdx.x;

  // Prefetch both batch rows of x while we build the filter spectrum.
  __builtin_prefetch(x + ((size_t)0 * 1024 + d) * 8192 + tid * 32, 0, 1);
  __builtin_prefetch(x + ((size_t)1 * 1024 + d) * 8192 + tid * 32, 0, 1);

  // DFT-128 matrix: D[k][a] = exp(-2*pi*i*((k*a) mod 128)/128). Symmetric:
  // row-major reads serve A-fragments, the same array serves B-fragments.
  for (int i = tid; i < 128 * 128; i += 256) {
    const int p = ((i >> 7) * (i & 127)) & 127;
    float s, c;
    sincosf(-(2.0f * PI_F / 128.0f) * (float)p, &s, &c);
    sDre[i] = (_Float16)c;
    sDim[i] = (_Float16)s;
  }
  // k[d] staged transposed: At[b][a] = k[128a+b]; zero rows a>=64 never read.
  const _Float16* krow = kf + (size_t)d * 8192;
  for (int i = tid; i < 8192; i += 256)
    sAt[(i & 127) * 64 + (i >> 7)] = krow[i];
  __syncthreads();

  // Forward FFT of k -> sK: G = tw.*(D*A) (A real, K=64); X = G*D.
  wave_cgemm<2>(sDre, sDim, 128, sAt, nullptr, 64, sGre, sGim, 128, false,
                8, 1.f, 1.f, 1.f, -1.f);
  __syncthreads();
  wave_cgemm<4>(sGre, sGim, 128, sDre, sDim, 128, sKre, sKim, 128, false,
                8, 1.f, 1.f, 1.f, 0.f);
  __syncthreads();

  for (int b = 0; b < 2; ++b) {
    const float* xr = x + ((size_t)b * 1024 + d) * 8192;
    for (int i = tid; i < 8192; i += 256)
      sAt[(i & 127) * 64 + (i >> 7)] = (_Float16)xr[i];
    __syncthreads();
    // Forward FFT of x -> sX
    wave_cgemm<2>(sDre, sDim, 128, sAt, nullptr, 64, sGre, sGim, 128, false,
                  8, 1.f, 1.f, 1.f, -1.f);
    __syncthreads();
    wave_cgemm<4>(sGre, sGim, 128, sDre, sDim, 128, sXre, sXim, 128, false,
                  8, 1.f, 1.f, 1.f, 0.f);
    __syncthreads();
    // Pointwise Y = X * Kf / 16384 (f32 math, v8h moves)
    for (int i = tid; i < 2048; i += 256) {
      v8h ar8 = ((v8h*)sXre)[i], ai8 = ((v8h*)sXim)[i];
      v8h kr8 = ((v8h*)sKre)[i], ki8 = ((v8h*)sKim)[i];
#pragma unroll
      for (int e = 0; e < 8; ++e) {
        const float ar = (float)ar8[e], ai = (float)ai8[e];
        const float kr = (float)kr8[e], ki = (float)ki8[e];
        ar8[e] = (_Float16)((ar * kr - ai * ki) * (1.0f / 16384.0f));
        ai8[e] = (_Float16)((ar * ki + ai * kr) * (1.0f / 16384.0f));
      }
      ((v8h*)sXre)[i] = ar8;
      ((v8h*)sXim)[i] = ai8;
    }
    __syncthreads();
    // Inverse: H = conj-tw.*(Y*conj(D)), stored TRANSPOSED (v8h stores; the
    // exact layout stage 2's B operand needs); z = Re(conj(D)*H'), rows < 64.
    wave_cgemm<4>(sXre, sXim, 128, sDre, sDim, 128, sGre, sGim, 128, true,
                  8, 1.f, -1.f, 1.f, +1.f);
    __syncthreads();
    wave_cgemm<4>(sDre, sDim, 128, sGre, sGim, 128, sXre, nullptr, 128, false,
                  4, -1.f, 1.f, 1.f, 0.f);
    __syncthreads();
    // y[n] + residual x*D (row-major [mu][nu] == linear n)
    const float Dd = Dvec[d];
    float* orow = out + ((size_t)b * 1024 + d) * 8192;
    for (int i = tid; i < 1024; i += 256) {
      const int j = i << 3;
      const v8h y8 = ((v8h*)sXre)[i];
#pragma unroll
      for (int e = 0; e < 8; ++e) orow[j + e] = (float)y8[e] + xr[j + e] * Dd;
    }
    __syncthreads();
  }
}

extern "C" void kernel_launch(void* const* d_in, const int* in_sizes, int n_in,
                              void* d_out, int out_size, void* d_ws,
                              size_t ws_size, hipStream_t stream) {
  (void)in_sizes; (void)n_in; (void)out_size; (void)ws_size;
  const float* x      = (const float*)d_in[0];
  // d_in[1] is L (= 8192, compile-time constant here)
  const float* z      = (const float*)d_in[2];
  const float* tpos   = (const float*)d_in[3];
  const float* deltas = (const float*)d_in[4];
  const float* freq   = (const float*)d_in[5];
  const float* W0     = (const float*)d_in[6];
  const float* b0     = (const float*)d_in[7];
  const float* W1     = (const float*)d_in[8];
  const float* b1     = (const float*)d_in[9];
  const float* W2     = (const float*)d_in[10];
  const float* b2     = (const float*)d_in[11];
  const float* Wout   = (const float*)d_in[12];
  const float* Dvec   = (const float*)d_in[13];
  float* out = (float*)d_out;
  _Float16* kfilt = (_Float16*)d_ws;  // 1024 x 8192 f16 = 16 MB

  hipLaunchKernelGGL(hyena_filter_kernel, dim3(64), dim3(256), 0, stream,
                     z, tpos, deltas, freq, W0, b0, W1, b1, W2, b2, Wout,
                     kfilt);
  hipLaunchKernelGGL(hyena_fftconv_kernel, dim3(1024), dim3(256), 0, stream,
                     x, kfilt, Dvec, out);
}
